// HetAgg_77043123356175
// MI455X (gfx1250) — compile-verified, matrix operand order checked by reading, not run on hardware
//
#include <hip/hip_runtime.h>
#include <stdint.h>

#define H 128
#define S 10
#define BATCH 16384

typedef __attribute__((ext_vector_type(16))) __bf16 v16bf;
typedef __attribute__((ext_vector_type(8)))  float  v8f;
typedef __attribute__((ext_vector_type(4))) unsigned int u32x4;
typedef __attribute__((ext_vector_type(4))) int i32x4;
typedef __attribute__((ext_vector_type(8))) int i32x8;

union FragB16 { v16bf v; uint32_t u[8]; };

__device__ __forceinline__ uint16_t f32_to_bf16(float f) {
  union { float f; uint32_t u; } x; x.f = f;
  uint32_t r = x.u + 0x7FFFu + ((x.u >> 16) & 1u);
  return (uint16_t)(r >> 16);
}

__device__ __forceinline__ v8f splat8(float x) {
  v8f r = {x, x, x, x, x, x, x, x};
  return r;
}

// ---- Tensor Data Mover helpers (CDNA5 ISA ch. 7/8, D# descriptor) ----------
// clang-23 toolchain: 6-arg builtin (g0 v4u, g1 v8i, g2 v4i, g3 v4i, v8i, cpol).
// 1-D linear load of n8 8-byte elements from global to LDS.
__device__ __forceinline__ void tdm_load_lin(uint32_t lds_off, const void* gptr,
                                             uint32_t n8) {
  uint64_t ga = (uint64_t)(uintptr_t)gptr;
  u32x4 g0 = {1u,                                   // count=1 (valid D#)
              lds_off,                              // lds_addr (bytes)
              (uint32_t)ga,                         // global_addr[31:0]
              (uint32_t)((ga >> 32) & 0x1FFFFFFu) | (2u << 30)};  // addr[56:32] | type=2
  i32x8 g1;
  g1[0] = (int)(3u << 16);                          // data_size = 8B
  g1[1] = (int)((n8 & 0xFFFFu) << 16);              // tensor_dim0[15:0]
  g1[2] = (int)(((n8 >> 16) & 0xFFFFu) | (1u << 16));  // dim0[31:16] | tensor_dim1=1
  g1[3] = (int)((n8 & 0xFFFFu) << 16);              // tile_dim0 = n8
  g1[4] = 0;                                        // tile_dim1/2 unused
  g1[5] = (int)n8;                                  // tensor_dim0_stride
  g1[6] = 0;
  g1[7] = 0;
  i32x4 z4 = {0, 0, 0, 0};
  i32x8 z8 = {0, 0, 0, 0, 0, 0, 0, 0};
  __builtin_amdgcn_tensor_load_to_lds(g0, g1, z4, z4, z8, 0);
}

// Gather-mode load: 16 rows of row8 8-byte elements each, rows selected by
// 16-bit indices packed in pk[0..7]; compacted into LDS sequentially.
__device__ __forceinline__ void tdm_gather16(uint32_t lds_off, const void* gptr,
                                             uint32_t row8, uint32_t nrows,
                                             const uint32_t* pk) {
  uint64_t ga = (uint64_t)(uintptr_t)gptr;
  u32x4 g0 = {1u | (1u << 31),                      // count=1 | gather_mode (16-bit idx)
              lds_off,
              (uint32_t)ga,
              (uint32_t)((ga >> 32) & 0x1FFFFFFu) | (2u << 30)};
  i32x8 g1;
  g1[0] = (int)(3u << 16);                          // data_size = 8B
  g1[1] = (int)((row8 & 0xFFFFu) << 16);            // tensor_dim0 = row length
  g1[2] = (int)(((row8 >> 16) & 0xFFFFu) | ((nrows & 0xFFFFu) << 16));  // | tensor_dim1
  g1[3] = (int)(((nrows >> 16) & 0xFFFFu) | ((row8 & 0xFFFFu) << 16));  // | tile_dim0
  g1[4] = 16;                                       // tile_dim1 = #valid indices
  g1[5] = (int)row8;                                // row stride (data_size units)
  g1[6] = 0;
  g1[7] = 0;
  i32x4 g2 = {(int)pk[0], (int)pk[1], (int)pk[2], (int)pk[3]};
  i32x4 g3 = {(int)pk[4], (int)pk[5], (int)pk[6], (int)pk[7]};
  i32x8 z8 = {0, 0, 0, 0, 0, 0, 0, 0};
  __builtin_amdgcn_tensor_load_to_lds(g0, g1, g2, g3, z8, 0);
}

// Load a 16x32 bf16 WMMA operand fragment from a row-major bf16 buffer.
__device__ __forceinline__ v16bf load_frag(const uint16_t* p, int ld, int row0, int kbase) {
  const int lane = threadIdx.x & 31;
  const int r = row0 + (lane & 15);
  const int half = (lane >> 4) & 1;
  const uint32_t* q = (const uint32_t*)(p + (size_t)r * ld);
  FragB16 f;
#pragma unroll
  for (int v = 0; v < 8; ++v) {
    int k = kbase + ((v < 4) ? (half * 8 + 2 * v) : (16 + half * 8 + 2 * (v - 4)));
    f.u[v] = q[k >> 1];
  }
  return f.v;
}

__global__ void cvt_kernel(const float* __restrict__ src, uint16_t* __restrict__ dst, int n) {
  int i = blockIdx.x * blockDim.x + threadIdx.x;
  if (i < n) dst[i] = f32_to_bf16(src[i]);
}

// Gather + project: out[r, :] = feats[idx[r], :] @ W^T + bias.
// Workgroup = 16 rows; TDM gather stages the 16 table rows, TDM linear load
// stages the whole weight matrix; 4 waves x 2 N-tiles of WMMA.
__global__ __launch_bounds__(128) void proj_kernel(
    const uint16_t* __restrict__ feats,   // [nrows, D] bf16
    const int* __restrict__ idx,          // [R]
    const uint16_t* __restrict__ W,       // [128, D] bf16 (row = output)
    const float* __restrict__ bias,       // [128]
    uint16_t* __restrict__ out_bf,        // [R, 128] or null
    float* __restrict__ out_f32,          // [R, 128] or null
    int D, int nrows) {
  __shared__ __align__(16) uint16_t At[16 * 512];    //  16 KB
  __shared__ __align__(16) uint16_t sW[128 * 512];   // 128 KB
  const int tid = threadIdx.x;

  if (tid < 32) {  // wave 0 issues the DMAs (EXEC-independent, TENSORcnt-tracked)
    const int* ip = idx + blockIdx.x * 16;
    uint32_t pk[8];
#pragma unroll
    for (int j = 0; j < 8; ++j)
      pk[j] = ((uint32_t)ip[2 * j] & 0xFFFFu) | ((uint32_t)ip[2 * j + 1] << 16);
    tdm_gather16((uint32_t)(size_t)At, feats, (uint32_t)(D >> 2), (uint32_t)nrows, pk);
    tdm_load_lin((uint32_t)(size_t)sW, W, (uint32_t)(128 * D) >> 2);
    __builtin_amdgcn_s_wait_tensorcnt((short)0);
  }
  __syncthreads();

  const int wv = tid >> 5, lane = tid & 31;
  const int kch = D >> 5;
  const size_t row0 = (size_t)blockIdx.x * 16;
#pragma unroll
  for (int u = 0; u < 2; ++u) {
    const int n0 = wv * 32 + u * 16;
    v8f c = splat8(bias[n0 + (lane & 15)]);
    for (int kc = 0; kc < kch; ++kc) {
      v16bf a = load_frag(At, D, 0, kc * 32);
      v16bf b = load_frag(sW, D, n0, kc * 32);
      c = __builtin_amdgcn_wmma_f32_16x16x32_bf16(false, a, false, b, (short)0, c, false, false);
    }
    const int col = n0 + (lane & 15);
#pragma unroll
    for (int v = 0; v < 8; ++v) {
      int m = v + 8 * (lane >> 4);
      size_t o = (row0 + m) * H + col;
      if (out_f32) out_f32[o] = c[v];
      if (out_bf)  out_bf[o] = f32_to_bf16(c[v]);
    }
  }
}

// GRU over S=10 steps, mean of hidden states. Workgroup = 16 rows.
// X tile + both 384x128 bf16 weight matrices staged by TDM and LDS-stationary.
__global__ __launch_bounds__(128) void gru_kernel(
    const uint16_t* __restrict__ X,      // [B, S, H] bf16
    const uint16_t* __restrict__ Wih,    // [3H, H] bf16
    const uint16_t* __restrict__ Whh,    // [3H, H] bf16
    const float* __restrict__ bih,       // [3H]
    const float* __restrict__ bhh,       // [3H]
    float* __restrict__ agg) {           // [B, H]
  __shared__ __align__(16) uint16_t Xt[16 * S * H];   // 40 KB
  __shared__ __align__(16) uint16_t hb[16 * H];       //  4 KB
  __shared__ __align__(16) uint16_t sWih[3 * H * H];  // 96 KB
  __shared__ __align__(16) uint16_t sWhh[3 * H * H];  // 96 KB

  const int tid = threadIdx.x;
  if (tid < 32) {
    tdm_load_lin((uint32_t)(size_t)Xt, X + (size_t)blockIdx.x * 16 * S * H,
                 (uint32_t)(16 * S * H * 2) >> 3);
    tdm_load_lin((uint32_t)(size_t)sWih, Wih, (uint32_t)(3 * H * H * 2) >> 3);
    tdm_load_lin((uint32_t)(size_t)sWhh, Whh, (uint32_t)(3 * H * H * 2) >> 3);
    __builtin_amdgcn_s_wait_tensorcnt((short)0);
  }
  for (int t = tid; t < 16 * H / 2; t += 128) ((uint32_t*)hb)[t] = 0u;
  __syncthreads();

  const int wv = tid >> 5, lane = tid & 31;
  const int jb = wv * 32;  // this wave owns output columns [jb, jb+32)
  v8f hreg[2], hsum[2];
  hreg[0] = hreg[1] = hsum[0] = hsum[1] = splat8(0.f);

  for (int s = 0; s < S; ++s) {
    v8f gi[3][2], gh[3][2];
#pragma unroll
    for (int g = 0; g < 3; ++g)
#pragma unroll
      for (int u = 0; u < 2; ++u) {
        int n = g * H + jb + u * 16 + (lane & 15);
        gi[g][u] = splat8(bih[n]);
        gh[g][u] = splat8(bhh[n]);
      }
#pragma unroll
    for (int kc = 0; kc < 4; ++kc) {
      v16bf ax = load_frag(Xt + s * H, S * H, 0, kc * 32);
      v16bf ah = load_frag(hb, H, 0, kc * 32);
#pragma unroll
      for (int g = 0; g < 3; ++g)
#pragma unroll
        for (int u = 0; u < 2; ++u) {
          const int n0 = g * H + jb + u * 16;
          v16bf bi = load_frag(sWih, H, n0, kc * 32);
          gi[g][u] = __builtin_amdgcn_wmma_f32_16x16x32_bf16(false, ax, false, bi, (short)0, gi[g][u], false, false);
          v16bf bh = load_frag(sWhh, H, n0, kc * 32);
          gh[g][u] = __builtin_amdgcn_wmma_f32_16x16x32_bf16(false, ah, false, bh, (short)0, gh[g][u], false, false);
        }
    }
    // gate math on C fragments (same (M,N) slot across r/z/n tiles)
#pragma unroll
    for (int u = 0; u < 2; ++u)
#pragma unroll
      for (int v = 0; v < 8; ++v) {
        float r = 1.f / (1.f + __expf(-(gi[0][u][v] + gh[0][u][v])));
        float z = 1.f / (1.f + __expf(-(gi[1][u][v] + gh[1][u][v])));
        float n = tanhf(gi[2][u][v] + r * gh[2][u][v]);
        float hn = (1.f - z) * n + z * hreg[u][v];
        hreg[u][v] = hn;
        hsum[u][v] += hn;
      }
    __syncthreads();  // everyone done reading old h
#pragma unroll
    for (int u = 0; u < 2; ++u) {
      const int col = jb + u * 16 + (lane & 15);
#pragma unroll
      for (int v = 0; v < 8; ++v) {
        int m = v + 8 * (lane >> 4);
        hb[m * H + col] = f32_to_bf16(hreg[u][v]);
      }
    }
    __syncthreads();  // new h visible to all waves
  }

  const size_t row0 = (size_t)blockIdx.x * 16;
#pragma unroll
  for (int u = 0; u < 2; ++u) {
    const int col = jb + u * 16 + (lane & 15);
#pragma unroll
    for (int v = 0; v < 8; ++v) {
      int m = v + 8 * (lane >> 4);
      agg[(row0 + m) * H + col] = hsum[u][v] * (1.f / S);
    }
  }
}

// 4-candidate attention + elu. One wave32 per row, shuffle reductions.
__global__ __launch_bounds__(256) void attn_kernel(
    const float* __restrict__ center, const float* __restrict__ g0,
    const float* __restrict__ g1, const float* __restrict__ g2,
    const float* __restrict__ att, float* __restrict__ outp) {
  const int wv = threadIdx.x >> 5, lane = threadIdx.x & 31;
  const size_t b = (size_t)blockIdx.x * 8 + wv;
  float cv[4], v0[4], v1[4], v2[4];
  float ds = 0.f, dc = 0.f, d0 = 0.f, d1 = 0.f, d2 = 0.f;
#pragma unroll
  for (int i = 0; i < 4; ++i) {
    int d = lane + 32 * i;
    float c = center[b * H + d];
    float x0 = g0[b * H + d], x1 = g1[b * H + d], x2 = g2[b * H + d];
    cv[i] = c; v0[i] = x0; v1[i] = x1; v2[i] = x2;
    float as = att[d], ac = att[H + d];
    ds += c * as; dc += c * ac; d0 += x0 * ac; d1 += x1 * ac; d2 += x2 * ac;
  }
#pragma unroll
  for (int off = 16; off > 0; off >>= 1) {
    ds += __shfl_xor(ds, off, 32);
    dc += __shfl_xor(dc, off, 32);
    d0 += __shfl_xor(d0, off, 32);
    d1 += __shfl_xor(d1, off, 32);
    d2 += __shfl_xor(d2, off, 32);
  }
  float e0 = ds + dc; e0 = e0 > 0.f ? e0 : 0.01f * e0;
  float e1 = ds + d0; e1 = e1 > 0.f ? e1 : 0.01f * e1;
  float e2 = ds + d1; e2 = e2 > 0.f ? e2 : 0.01f * e2;
  float e3 = ds + d2; e3 = e3 > 0.f ? e3 : 0.01f * e3;
  float mx = fmaxf(fmaxf(e0, e1), fmaxf(e2, e3));
  float p0 = __expf(e0 - mx), p1 = __expf(e1 - mx), p2 = __expf(e2 - mx), p3 = __expf(e3 - mx);
  float inv = 1.f / (p0 + p1 + p2 + p3);
  p0 *= inv; p1 *= inv; p2 *= inv; p3 *= inv;
#pragma unroll
  for (int i = 0; i < 4; ++i) {
    int d = lane + 32 * i;
    float y = p0 * cv[i] + p1 * v0[i] + p2 * v1[i] + p3 * v2[i];
    outp[b * H + d] = y > 0.f ? y : (__expf(y) - 1.f);
  }
}

extern "C" void kernel_launch(void* const* d_in, const int* in_sizes, int n_in,
                              void* d_out, int out_size, void* d_ws, size_t ws_size,
                              hipStream_t stream) {
  const float* cell_feats = (const float*)d_in[0];
  const float* drug_feats = (const float*)d_in[1];
  const float* gene_feats = (const float*)d_in[2];
  const float* W_cell = (const float*)d_in[3];
  const float* b_cell = (const float*)d_in[4];
  const float* W_drug = (const float*)d_in[5];
  const float* b_drug = (const float*)d_in[6];
  const float* W_gene = (const float*)d_in[7];
  const float* b_gene = (const float*)d_in[8];
  const float* gru_wih_f = (const float*)d_in[9];
  const float* gru_whh_f = (const float*)d_in[10];
  const float* gru_bih = (const float*)d_in[11];
  const float* gru_bhh = (const float*)d_in[12];
  const float* att_a = (const float*)d_in[13];
  const int* center_idx = (const int*)d_in[14];
  const int* neigh_cell = (const int*)d_in[15];
  const int* neigh_drug = (const int*)d_in[16];
  const int* neigh_gene = (const int*)d_in[17];

  char* ws = (char*)d_ws;
  size_t off = 0;
  auto alloc = [&](size_t bytes) -> void* {
    void* p = ws + off;
    off = (off + bytes + 255) & ~(size_t)255;
    return p;
  };

  uint16_t* tblc = (uint16_t*)alloc((size_t)2000 * 128 * 2);
  uint16_t* tbld = (uint16_t*)alloc((size_t)4000 * 512 * 2);
  uint16_t* tblg = (uint16_t*)alloc((size_t)20000 * 256 * 2);
  uint16_t* wc   = (uint16_t*)alloc((size_t)128 * 128 * 2);
  uint16_t* wd   = (uint16_t*)alloc((size_t)128 * 512 * 2);
  uint16_t* wg   = (uint16_t*)alloc((size_t)128 * 256 * 2);
  uint16_t* wih  = (uint16_t*)alloc((size_t)2 * 3 * 384 * 128 * 2);
  uint16_t* whh  = (uint16_t*)alloc((size_t)2 * 3 * 384 * 128 * 2);
  uint16_t* Xbf  = (uint16_t*)alloc((size_t)BATCH * S * H * 2);
  float* agg0 = (float*)alloc((size_t)BATCH * H * 4);
  float* agg1 = (float*)alloc((size_t)BATCH * H * 4);
  float* agg2 = (float*)alloc((size_t)BATCH * H * 4);
  float* cA   = (float*)alloc((size_t)BATCH * H * 4);
  float* cB   = (float*)alloc((size_t)BATCH * H * 4);

  auto cvt = [&](const float* s, uint16_t* d, int n) {
    cvt_kernel<<<(n + 255) / 256, 256, 0, stream>>>(s, d, n);
  };
  cvt(cell_feats, tblc, 2000 * 128);
  cvt(drug_feats, tbld, 4000 * 512);
  cvt(gene_feats, tblg, 20000 * 256);
  cvt(W_cell, wc, 128 * 128);
  cvt(W_drug, wd, 128 * 512);
  cvt(W_gene, wg, 128 * 256);
  cvt(gru_wih_f, wih, 2 * 3 * 384 * 128);
  cvt(gru_whh_f, whh, 2 * 3 * 384 * 128);

  // center = proj(drug_feats[center_idx]) in f32
  proj_kernel<<<BATCH / 16, 128, 0, stream>>>(tbld, center_idx, wd, b_drug,
                                              nullptr, cA, 512, 4000);

  const uint16_t* tbls[3] = {tblc, tbld, tblg};
  const uint16_t* wts[3] = {wc, wd, wg};
  const float* bs[3] = {b_cell, b_drug, b_gene};
  const int* neigh[3] = {neigh_cell, neigh_drug, neigh_gene};
  const int Ds[3] = {128, 512, 256};
  const int NR[3] = {2000, 4000, 20000};
  const int RN = BATCH * S;  // 163840 neighbor rows per (layer, type)
  float* aggs[3] = {agg0, agg1, agg2};

  for (int l = 0; l < 2; ++l) {
    for (int t = 0; t < 3; ++t) {
      proj_kernel<<<RN / 16, 128, 0, stream>>>(
          tbls[t], neigh[t] + (size_t)l * RN, wts[t], bs[t], Xbf, nullptr,
          Ds[t], NR[t]);
      const size_t wo = (size_t)(l * 3 + t) * 384 * 128;
      gru_kernel<<<BATCH / 16, 128, 0, stream>>>(
          Xbf, wih + wo, whh + wo,
          gru_bih + (l * 3 + t) * 384, gru_bhh + (l * 3 + t) * 384, aggs[t]);
    }
    const float* cin = (l == 0) ? cA : cB;
    float* cout = (l == 1) ? (float*)d_out : cB;
    attn_kernel<<<BATCH / 8, 256, 0, stream>>>(cin, agg0, agg1, agg2,
                                               att_a + l * 256, cout);
  }
}
